// PerStepBLSTM_15229954031737
// MI455X (gfx1250) — compile-verified, matrix operand order checked by reading, not run on hardware
//
#include <hip/hip_runtime.h>
#include <hip/hip_bf16.h>

// ---------------------------------------------------------------------------
// Bidirectional LSTM, B=32, T=256, E=H=512 (4H = 2048 gates).
//
// Strategy:
//   * One-time prep kernels (per launch): repack x -> [T][B][E] bf16,
//     build combined weight W[dir][4H][1024] bf16 where rows are
//     [Whh ; Wih] transposed (column-of-gates major, K contiguous),
//     combine biases, zero h0/c0.
//   * 256 sequential step kernels: g = [h_prev | x_t] @ Wcomb + b  via
//     v_wmma_f32_16x16x32_bf16, then fused sigmoid/tanh cell update.
//     h ping-pongs between two bf16 buffers in workspace; c stays f32.
// ---------------------------------------------------------------------------

typedef __attribute__((ext_vector_type(16))) __bf16 v16bf;
typedef __attribute__((ext_vector_type(8)))  float  v8f;

#define Bc 32
#define Tc 256
#define Ec 512
#define Hc 512
#define G4 2048   // 4*H
#define Kc 1024   // H + E (fused K)

union Frag {
    v16bf v;
    uint4 q[2];
};

__device__ __forceinline__ float sigf(float x) {
    return 1.0f / (1.0f + __expf(-x));
}
__device__ __forceinline__ float tanhfast(float x) {
    return 2.0f * sigf(2.0f * x) - 1.0f;
}

// ---------------------------------------------------------------------------
// Prep: x [B][T][E] f32  ->  xpack [T][B][E] bf16
// ---------------------------------------------------------------------------
__global__ __launch_bounds__(256) void blstm_pack_x(
    const float* __restrict__ x, __hip_bfloat16* __restrict__ xpack)
{
    int tid = blockIdx.x * 256 + threadIdx.x;        // 0 .. B*T*E-1 (dst index)
    int e = tid & (Ec - 1);
    int b = (tid >> 9) & (Bc - 1);
    int t = tid >> 14;                                // /(B*E)
    xpack[tid] = __float2bfloat16(x[((b * Tc) + t) * Ec + e]);
}

// ---------------------------------------------------------------------------
// Prep: Wcomb[dir][n][k] bf16, n in [0,4H), k in [0,1024):
//   k <  512 : Whh[k][n]        (hidden part first, matches [h | x] order)
//   k >= 512 : Wih[k-512][n]
// ---------------------------------------------------------------------------
__global__ __launch_bounds__(256) void blstm_pack_w(
    const float* __restrict__ fWih, const float* __restrict__ fWhh,
    const float* __restrict__ bWih, const float* __restrict__ bWhh,
    __hip_bfloat16* __restrict__ Wcomb)
{
    int tid = blockIdx.x * 256 + threadIdx.x;        // 0 .. 2*2048*1024-1
    int k   = tid & (Kc - 1);
    int n   = (tid >> 10) & (G4 - 1);
    int dir = tid >> 21;
    const float* Wih = dir ? bWih : fWih;
    const float* Whh = dir ? bWhh : fWhh;
    float v = (k < Hc) ? Whh[k * G4 + n] : Wih[(k - Hc) * G4 + n];
    Wcomb[tid] = __float2bfloat16(v);
}

// ---------------------------------------------------------------------------
// Prep: bcomb[dir][4H] = bih + bhh ; zero h0 (ping buffer 0) and c
// ---------------------------------------------------------------------------
__global__ __launch_bounds__(256) void blstm_init(
    const float* __restrict__ fbih, const float* __restrict__ fbhh,
    const float* __restrict__ bbih, const float* __restrict__ bbhh,
    float* __restrict__ bcomb, __hip_bfloat16* __restrict__ hbuf,
    float* __restrict__ cbuf)
{
    int tid = blockIdx.x * 256 + threadIdx.x;        // 0 .. 32767
    if (tid < 2 * G4) {
        int dir = tid >> 11;
        int n   = tid & (G4 - 1);
        bcomb[tid] = dir ? (bbih[n] + bbhh[n]) : (fbih[n] + fbhh[n]);
    }
    hbuf[tid] = __float2bfloat16(0.0f);              // pp=0, both dirs: 2*32*512
    cbuf[tid] = 0.0f;                                // both dirs: 2*32*512
}

// ---------------------------------------------------------------------------
// One recurrence step. grid = (16 col-groups, 2 dirs), block = 256 (8 waves).
// Each workgroup: 32 h-columns (n0..n0+31), all 4 gates.
//   wave w: gate g=w>>1, mtile=w&1 (rows of B), ntiles 0 and 1.
//   K loop: phase 0 reads h_prev (k 0..511), phase 1 reads x_t (k 0..511).
// ---------------------------------------------------------------------------
__global__ __launch_bounds__(256) void blstm_step(
    const __hip_bfloat16* __restrict__ xpack,
    const __hip_bfloat16* __restrict__ Wcomb,
    const float* __restrict__ bcomb,
    __hip_bfloat16* __restrict__ hbuf,
    float* __restrict__ cbuf,
    float* __restrict__ out,
    int t)
{
    __shared__ float gshare[4][32][32];              // 16 KB gate staging

    const int dir  = blockIdx.y;
    const int n0   = blockIdx.x * 32;
    const int tid  = threadIdx.x;
    const int w    = tid >> 5;
    const int lane = tid & 31;
    const int kh   = lane >> 4;                      // K-half select (wave32 WMMA layout)
    const int lm   = lane & 15;
    const int g     = w >> 1;                        // gate 0..3 (i,f,gc,o)
    const int mtile = w & 1;                         // B-row tile
    const int ppr   = t & 1;                         // h read buffer
    const int ppw   = (t + 1) & 1;                   // h write buffer
    const int t_eff = dir ? (Tc - 1 - t) : t;        // bwd processes reversed input

    const int m = mtile * 16 + lm;                   // A-matrix row (batch index)
    const __hip_bfloat16* aRowH = hbuf  + (size_t)(((ppr * 2 + dir) * Bc + m) * Hc);
    const __hip_bfloat16* aRowX = xpack + (size_t)((t_eff * Bc + m) * Ec);

    const int col0 = g * Hc + n0 + lm;               // B-matrix column (gate space)
    const __hip_bfloat16* bRow0 =
        Wcomb + (size_t)dir * G4 * Kc + (size_t)col0 * Kc;
    const __hip_bfloat16* bRow1 = bRow0 + (size_t)16 * Kc;

    // Accumulators initialized with the (column-only) combined bias.
    const float bias0 = bcomb[dir * G4 + g * Hc + n0 + lm];
    const float bias1 = bcomb[dir * G4 + g * Hc + n0 + 16 + lm];
    v8f acc0, acc1;
#pragma unroll
    for (int r = 0; r < 8; ++r) { acc0[r] = bias0; acc1[r] = bias1; }

#pragma unroll
    for (int p = 0; p < 2; ++p) {
        const __hip_bfloat16* aRow = p ? aRowX : aRowH;
        const __hip_bfloat16* b0   = bRow0 + p * Hc;
        const __hip_bfloat16* b1   = bRow1 + p * Hc;
#pragma unroll 4
        for (int kc = 0; kc < 16; ++kc) {
            const int e0 = kc * 32 + kh * 8;         // bf16 element offset, 16B aligned
            Frag a, fb0, fb1;
            a.q[0]   = *(const uint4*)(aRow + e0);
            a.q[1]   = *(const uint4*)(aRow + e0 + 16);
            fb0.q[0] = *(const uint4*)(b0 + e0);
            fb0.q[1] = *(const uint4*)(b0 + e0 + 16);
            fb1.q[0] = *(const uint4*)(b1 + e0);
            fb1.q[1] = *(const uint4*)(b1 + e0 + 16);
            acc0 = __builtin_amdgcn_wmma_f32_16x16x32_bf16(
                false, a.v, false, fb0.v, (short)0, acc0, false, false);
            acc1 = __builtin_amdgcn_wmma_f32_16x16x32_bf16(
                false, a.v, false, fb1.v, (short)0, acc1, false, false);
        }
    }

    // Scatter gate tiles to LDS: C/D layout -> lane kh selects M-half.
#pragma unroll
    for (int r = 0; r < 8; ++r) {
        const int M = mtile * 16 + kh * 8 + r;
        gshare[g][M][lm]      = acc0[r];
        gshare[g][M][16 + lm] = acc1[r];
    }
    __syncthreads();

    // Fused elementwise LSTM update: 32 rows x 32 cols = 1024 outputs.
#pragma unroll
    for (int j = 0; j < 4; ++j) {
        const int idx = tid + 256 * j;
        const int b   = idx >> 5;
        const int nn  = idx & 31;
        const int col = n0 + nn;
        const float iv = gshare[0][b][nn];
        const float fv = gshare[1][b][nn];
        const float gv = gshare[2][b][nn];
        const float ov = gshare[3][b][nn];
        const size_t cidx = (size_t)((dir * Bc + b) * Hc + col);
        const float cprev = cbuf[cidx];
        const float cnew  = sigf(fv) * cprev + sigf(iv) * tanhfast(gv);
        const float hnew  = sigf(ov) * tanhfast(cnew);
        cbuf[cidx] = cnew;
        hbuf[(size_t)(((ppw * 2 + dir) * Bc + b) * Hc + col)] =
            __float2bfloat16(hnew);
        out[(size_t)((b * Tc + t) * (2 * Hc) + dir * Hc + col)] = hnew;
    }
}

// ---------------------------------------------------------------------------
extern "C" void kernel_launch(void* const* d_in, const int* in_sizes, int n_in,
                              void* d_out, int out_size, void* d_ws, size_t ws_size,
                              hipStream_t stream)
{
    (void)in_sizes; (void)n_in; (void)out_size; (void)ws_size;

    const float* x    = (const float*)d_in[0];
    const float* fWih = (const float*)d_in[1];
    const float* fbih = (const float*)d_in[2];
    const float* fWhh = (const float*)d_in[3];
    const float* fbhh = (const float*)d_in[4];
    const float* bWih = (const float*)d_in[5];
    const float* bbih = (const float*)d_in[6];
    const float* bWhh = (const float*)d_in[7];
    const float* bbhh = (const float*)d_in[8];
    float* out = (float*)d_out;

    // Workspace layout (bytes):
    //   [0, 8MiB)          xpack  bf16 [T][B][E]
    //   [8MiB, 16MiB)      Wcomb  bf16 [2][4H][1024]
    //   [16MiB, +64KiB)    bcomb  f32  [2][4H]
    //   next 128KiB        hbuf   bf16 [2pp][2dir][B][H]
    //   next 128KiB        cbuf   f32  [2dir][B][H]
    char* ws = (char*)d_ws;
    __hip_bfloat16* xpack = (__hip_bfloat16*)(ws);
    __hip_bfloat16* Wcomb = (__hip_bfloat16*)(ws + (size_t)(8u << 20));
    float*          bcomb = (float*)(ws + (size_t)(16u << 20));
    __hip_bfloat16* hbuf  = (__hip_bfloat16*)(ws + (size_t)(16u << 20) + (64u << 10));
    float*          cbuf  = (float*)(ws + (size_t)(16u << 20) + (64u << 10) + (128u << 10));

    // Prep (deterministic every call; also re-zeros h0/c0 state).
    blstm_pack_x<<<(Bc * Tc * Ec) / 256, 256, 0, stream>>>(x, xpack);
    blstm_pack_w<<<(2 * G4 * Kc) / 256, 256, 0, stream>>>(fWih, fWhh, bWih, bWhh, Wcomb);
    blstm_init<<<(2 * Bc * Hc) / 256, 256, 0, stream>>>(fbih, fbhh, bbih, bbhh,
                                                        bcomb, hbuf, cbuf);

    // Sequential recurrence: stream order enforces step dependencies.
    dim3 grid(Hc / 32, 2, 1);
    for (int t = 0; t < Tc; ++t) {
        blstm_step<<<grid, 256, 0, stream>>>(xpack, Wcomb, bcomb, hbuf, cbuf, out, t);
    }
}
